// UpdateLayer_24859270709887
// MI455X (gfx1250) — matrix operand order repaired and checked by reading, not compiled, not run on hardware
//
#include <hip/hip_runtime.h>

typedef unsigned int u32;
typedef unsigned short u16;
typedef __attribute__((ext_vector_type(16))) __bf16 v16bf;
typedef __attribute__((ext_vector_type(8)))  float  v8f;
typedef int v4i_ __attribute__((vector_size(16)));   // matches async builtin pointee

#define T_DIM 1024
#define B_DIM 4
#define E_DIM 512
#define H_DIM 8
#define HD_DIM 64
#define BH_DIM 32   // B*H
#define M_DIM 4096  // T*B

#define AS1 __attribute__((address_space(1)))
#define AS3 __attribute__((address_space(3)))

#if __has_builtin(__builtin_amdgcn_global_load_async_to_lds_b128)
#define HAVE_ASYNC_LDS 1
#else
#define HAVE_ASYNC_LDS 0
#endif

__device__ __forceinline__ void wait_async0() {
#if __has_builtin(__builtin_amdgcn_s_wait_asynccnt)
    __builtin_amdgcn_s_wait_asynccnt(0);
#else
    asm volatile("s_wait_asynccnt 0x0" ::: "memory");
#endif
}

union Frag { u32 u[8]; v16bf v; };

// f32 -> bf16 via hardware convert (RNE)
__device__ __forceinline__ u16 bf_bits(float f) {
    union { __bf16 h; u16 u; } x;
    x.h = (__bf16)f;
    return x.u;
}
__device__ __forceinline__ u32 pack2(float a, float b) {
    union { __bf16 h[2]; u32 u; } x;
    x.h[0] = (__bf16)a; x.h[1] = (__bf16)b;
    return x.u;
}

// 16-bit A-matrix 16x32 K map (contiguous order k = 16*half + s): VGPR i holds K pair base
__device__ __forceinline__ int akmap(int i, int half) {
    return 2 * i + 16 * half;
}

// load 16 contiguous bf16 (32B) as a B fragment (lane-local)
__device__ __forceinline__ Frag load_frag16(const u16* p) {
    Frag f;
    const uint4* q = (const uint4*)p;
    *(uint4*)&f.u[0] = q[0];
    *(uint4*)&f.u[4] = q[1];
    return f;
}

__device__ __forceinline__ v8f wmma_bf16(const Frag& a, const Frag& b, v8f c) {
    return __builtin_amdgcn_wmma_f32_16x16x32_bf16(false, a.v, false, b.v,
                                                   (short)0, c, false, false);
}

#define LDH 56  // LDS row stride (halfs): 112B (16B-aligned rows), conflict-free

// ---------------------------------------------------------------------------
// Kernel 0: elementwise f32 -> bf16 (query / W_in / W_out pre-conversion)
// ---------------------------------------------------------------------------
__global__ __launch_bounds__(256)
void cvt_bf16_kernel(const float* __restrict__ src, u16* __restrict__ dst, int n4) {
    int idx = blockIdx.x * 256 + threadIdx.x;
    if (idx < n4) {
        float4 f = ((const float4*)src)[idx];
        uint2 pk; pk.x = pack2(f.x, f.y); pk.y = pack2(f.z, f.w);
        ((uint2*)dst)[idx] = pk;
    }
}

// ---------------------------------------------------------------------------
// Kernel 1: qkv = query_bf @ Win_bf^T + bias ; scatter Q(*0.125)/K -> [BH][T][HD],
//           V -> transposed [BH][HD][T]. All-bf16 operands: async-DMA staging,
//           double-buffered LDS, 1 barrier per 32-deep k-step.
// ---------------------------------------------------------------------------
__global__ __launch_bounds__(256)
void qkv_proj_kernel(const u16* __restrict__ A, const u16* __restrict__ W,
                     const float* __restrict__ bias,
                     u16* __restrict__ q_ws, u16* __restrict__ k_ws,
                     u16* __restrict__ vT_ws) {
    __shared__ u16 a_lds[2][128 * LDH];
    __shared__ u16 b_lds[2][64 * LDH];
    const int tid = threadIdx.x;
    const int wave = tid >> 5, lane = tid & 31;
    const int half = lane >> 4, ln = lane & 15;
    const int wm = (wave >> 1) * 32, wn = (wave & 1) * 32;
    const int mb = blockIdx.x * 128, nb = blockIdx.y * 64;
    const int rr4 = tid >> 2, cg2 = tid & 3;  // row / 16B-chunk for staging

    v8f acc[2][2];
#pragma unroll
    for (int a = 0; a < 2; a++)
#pragma unroll
        for (int b = 0; b < 2; b++)
#pragma unroll
            for (int e = 0; e < 8; e++) acc[a][b][e] = 0.f;

#if HAVE_ASYNC_LDS
    auto stageA = [&](int buf, int k0) {
#pragma unroll
        for (int j = 0; j < 2; j++) {
            int row = rr4 + j * 64;
            u16* g = const_cast<u16*>(A) + (size_t)(mb + row) * E_DIM + k0 + cg2 * 8;
            u16* l = &a_lds[buf][row * LDH + cg2 * 8];
            __builtin_amdgcn_global_load_async_to_lds_b128((AS1 v4i_*)g, (AS3 v4i_*)l, 0, 0);
        }
    };
    auto stageB = [&](int buf, int k0) {
        u16* g = const_cast<u16*>(W) + (size_t)(nb + rr4) * E_DIM + k0 + cg2 * 8;
        u16* l = &b_lds[buf][rr4 * LDH + cg2 * 8];
        __builtin_amdgcn_global_load_async_to_lds_b128((AS1 v4i_*)g, (AS3 v4i_*)l, 0, 0);
    };
#else
    const int r8 = tid >> 3, cg = tid & 7;
    uint2 ua[4], ub[2];
    auto ldgAB = [&](int k0) {
#pragma unroll
        for (int p = 0; p < 4; p++)
            ua[p] = *(const uint2*)(A + (size_t)(mb + p * 32 + r8) * E_DIM + k0 + cg * 4);
#pragma unroll
        for (int p = 0; p < 2; p++)
            ub[p] = *(const uint2*)(W + (size_t)(nb + p * 32 + r8) * E_DIM + k0 + cg * 4);
    };
    auto cstAB = [&](int buf) {
#pragma unroll
        for (int p = 0; p < 4; p++)
            *(uint2*)&a_lds[buf][(p * 32 + r8) * LDH + cg * 4] = ua[p];
#pragma unroll
        for (int p = 0; p < 2; p++)
            *(uint2*)&b_lds[buf][(p * 32 + r8) * LDH + cg * 4] = ub[p];
    };
#endif

#if HAVE_ASYNC_LDS
    stageA(0, 0); stageB(0, 0);
#else
    ldgAB(0); cstAB(0);
#endif

    for (int kt = 0; kt < 16; kt++) {
        const int cur = kt & 1;
        const bool hn = kt < 15;
#if HAVE_ASYNC_LDS
        wait_async0();        // own DMA for buf[cur] complete
        __syncthreads();      // all waves' buf[cur] visible; buf[cur^1] readers retired
        if (hn) { stageA(cur ^ 1, (kt + 1) * 32); stageB(cur ^ 1, (kt + 1) * 32); }
#else
        if (hn) ldgAB((kt + 1) * 32);
        __syncthreads();
#endif

        Frag af[2], bf_[2];
#pragma unroll
        for (int a = 0; a < 2; a++)
#pragma unroll
            for (int i = 0; i < 8; i++)
                af[a].u[i] = *(const u32*)&a_lds[cur][(wm + a * 16 + ln) * LDH + akmap(i, half)];
#pragma unroll
        for (int b = 0; b < 2; b++)
#pragma unroll
            for (int i = 0; i < 8; i++)
                bf_[b].u[i] = *(const u32*)&b_lds[cur][(wn + b * 16 + ln) * LDH + half * 16 + 2 * i];
#pragma unroll
        for (int a = 0; a < 2; a++)
#pragma unroll
            for (int b = 0; b < 2; b++)
                acc[a][b] = wmma_bf16(af[a], bf_[b], acc[a][b]);

#if !HAVE_ASYNC_LDS
        if (hn) cstAB(cur ^ 1);
#endif
    }

    const int sec = (nb + wn) >> 9;  // uniform per block
#pragma unroll
    for (int b = 0; b < 2; b++) {
        int nn = nb + wn + b * 16 + ln;
        float bv = bias[nn];
        int rr = nn & 511, h = rr >> 6, hd = rr & 63;
#pragma unroll
        for (int a = 0; a < 2; a++) {
#pragma unroll
            for (int i = 0; i < 8; i++) {
                int m = mb + wm + a * 16 + half * 8 + i;
                float v = acc[a][b][i] + bv;
                int t = m >> 2, bq = m & 3;
                int bh = bq * H_DIM + h;
                if (sec == 0)
                    q_ws[((size_t)bh * T_DIM + t) * HD_DIM + hd] = bf_bits(v * 0.125f);
                else if (sec == 1)
                    k_ws[((size_t)bh * T_DIM + t) * HD_DIM + hd] = bf_bits(v);
                else
                    vT_ws[((size_t)bh * HD_DIM + hd) * T_DIM + t] = bf_bits(v);
            }
        }
    }
}

// ---------------------------------------------------------------------------
// Kernel 2: flash attention with additive bias from dist[B,T,T,H]
// grid (T/128, BH), 256 threads; each wave owns 16 query rows
// ---------------------------------------------------------------------------
__global__ __launch_bounds__(256)
void attn_kernel(const u16* __restrict__ q_ws, const u16* __restrict__ k_ws,
                 const u16* __restrict__ vT_ws, const float* __restrict__ dist,
                 u16* __restrict__ o_ws) {
    __shared__ u16 p_lds[8][16 * 40];  // per-wave P staging, stride 40 (conflict-free)
    const int tid = threadIdx.x;
    const int wave = tid >> 5, lane = tid & 31;
    const int half = lane >> 4, ln = lane & 15;
    const int bh = blockIdx.y, b = bh >> 3, h = bh & 7;
    const int tb = blockIdx.x * 128 + wave * 16;
    const float L2E = 1.44269504088896340736f;

    Frag qf[2];
    {
        const u16* qrow = q_ws + ((size_t)bh * T_DIM + tb + ln) * HD_DIM;
#pragma unroll
        for (int c = 0; c < 2; c++)
#pragma unroll
            for (int i = 0; i < 8; i++)
                qf[c].u[i] = *(const u32*)(qrow + c * 32 + akmap(i, half));
    }

    float m_i[8], l_i[8];
    v8f oa[4];
#pragma unroll
    for (int i = 0; i < 8; i++) { m_i[i] = -1e30f; l_i[i] = 0.f; }
#pragma unroll
    for (int t4 = 0; t4 < 4; t4++)
#pragma unroll
        for (int e = 0; e < 8; e++) oa[t4][e] = 0.f;

    for (int s0 = 0; s0 < T_DIM; s0 += 32) {
        // prefetch the bias stream one 64-key block ahead (128B line per lane)
        {
            int spf = s0 + 64;
            if (spf + ln * 4 < T_DIM) {
#pragma unroll
                for (int i = 0; i < 8; i++) {
                    const float* pp = dist
                        + ((size_t)(b * T_DIM + tb + half * 8 + i) * T_DIM + spf) * H_DIM
                        + (size_t)ln * 32;
                    __builtin_prefetch(pp, 0, 1);
                }
            }
        }

        // S = Q K^T  (two 16x16 n-tiles over 32 keys, HD=64 -> two k-steps)
        v8f st[2];
#pragma unroll
        for (int t2 = 0; t2 < 2; t2++)
#pragma unroll
            for (int e = 0; e < 8; e++) st[t2][e] = 0.f;
#pragma unroll
        for (int tile = 0; tile < 2; tile++) {
            const u16* krow = k_ws + ((size_t)bh * T_DIM + s0 + tile * 16 + ln) * HD_DIM;
#pragma unroll
            for (int c = 0; c < 2; c++) {
                Frag kf = load_frag16(krow + c * 32 + half * 16);
                st[tile] = wmma_bf16(qf[c], kf, st[tile]);
            }
        }

        // bias add + online softmax (row ops via 16-lane xor shuffles)
#pragma unroll
        for (int i = 0; i < 8; i++) {
            int t = tb + half * 8 + i;
            size_t bb = ((size_t)(b * T_DIM + t) * T_DIM + s0) * H_DIM + h;
            float v0 = st[0][i] + dist[bb + (size_t)ln * H_DIM];
            float v1 = st[1][i] + dist[bb + (size_t)(16 + ln) * H_DIM];
            float mx = fmaxf(v0, v1);
            mx = fmaxf(mx, __shfl_xor(mx, 1, 32));
            mx = fmaxf(mx, __shfl_xor(mx, 2, 32));
            mx = fmaxf(mx, __shfl_xor(mx, 4, 32));
            mx = fmaxf(mx, __shfl_xor(mx, 8, 32));
            float mnew = fmaxf(m_i[i], mx);
            float corr = exp2f((m_i[i] - mnew) * L2E);
            m_i[i] = mnew;
            float p0 = exp2f((v0 - mnew) * L2E);
            float p1 = exp2f((v1 - mnew) * L2E);
            float rs = p0 + p1;
            rs += __shfl_xor(rs, 1, 32);
            rs += __shfl_xor(rs, 2, 32);
            rs += __shfl_xor(rs, 4, 32);
            rs += __shfl_xor(rs, 8, 32);
            l_i[i] = l_i[i] * corr + rs;
#pragma unroll
            for (int t4 = 0; t4 < 4; t4++) oa[t4][i] *= corr;
            int r = half * 8 + i;
            p_lds[wave][r * 40 + ln] = bf_bits(p0);
            p_lds[wave][r * 40 + 16 + ln] = bf_bits(p1);
        }
        // wave-private LDS round trip: drain DS ops, then read P in A-layout
        asm volatile("s_wait_dscnt 0x0" ::: "memory");
        Frag pf;
#pragma unroll
        for (int i = 0; i < 8; i++)
            pf.u[i] = *(const u32*)&p_lds[wave][ln * 40 + akmap(i, half)];

        // O += P V  (V transposed: [BH][HD][T] -> contiguous K loads)
#pragma unroll
        for (int tile = 0; tile < 4; tile++) {
            const u16* vrow = vT_ws + ((size_t)bh * HD_DIM + tile * 16 + ln) * T_DIM
                              + s0 + half * 16;
            Frag vf = load_frag16(vrow);
            oa[tile] = wmma_bf16(pf, vf, oa[tile]);
        }
    }

    float linv[8];
#pragma unroll
    for (int i = 0; i < 8; i++) linv[i] = 1.0f / l_i[i];
#pragma unroll
    for (int tile = 0; tile < 4; tile++) {
#pragma unroll
        for (int i = 0; i < 8; i++) {
            float v = oa[tile][i] * linv[i];
            int t = tb + half * 8 + i;
            int e = h * HD_DIM + tile * 16 + ln;
            o_ws[((size_t)t * B_DIM + b) * E_DIM + e] = bf_bits(v);
        }
    }
}

// ---------------------------------------------------------------------------
// Kernel 3: out = o_ws @ Wout_bf^T + bout (f32 output). Both operands bf16:
// async-DMA staged, double-buffered.
// ---------------------------------------------------------------------------
__global__ __launch_bounds__(256)
void out_proj_kernel(const u16* __restrict__ Ao, const u16* __restrict__ W,
                     const float* __restrict__ bias, float* __restrict__ out) {
    __shared__ u16 a_lds[2][128 * LDH];
    __shared__ u16 b_lds[2][64 * LDH];
    const int tid = threadIdx.x;
    const int wave = tid >> 5, lane = tid & 31;
    const int half = lane >> 4, ln = lane & 15;
    const int wm = (wave >> 1) * 32, wn = (wave & 1) * 32;
    const int mb = blockIdx.x * 128, nb = blockIdx.y * 64;
    const int rr4 = tid >> 2, cg2 = tid & 3;

    v8f acc[2][2];
#pragma unroll
    for (int a = 0; a < 2; a++)
#pragma unroll
        for (int b = 0; b < 2; b++)
#pragma unroll
            for (int e = 0; e < 8; e++) acc[a][b][e] = 0.f;

#if HAVE_ASYNC_LDS
    auto stageA = [&](int buf, int k0) {
#pragma unroll
        for (int j = 0; j < 2; j++) {
            int row = rr4 + j * 64;
            u16* g = const_cast<u16*>(Ao) + (size_t)(mb + row) * E_DIM + k0 + cg2 * 8;
            u16* l = &a_lds[buf][row * LDH + cg2 * 8];
            __builtin_amdgcn_global_load_async_to_lds_b128((AS1 v4i_*)g, (AS3 v4i_*)l, 0, 0);
        }
    };
    auto stageB = [&](int buf, int k0) {
        u16* g = const_cast<u16*>(W) + (size_t)(nb + rr4) * E_DIM + k0 + cg2 * 8;
        u16* l = &b_lds[buf][rr4 * LDH + cg2 * 8];
        __builtin_amdgcn_global_load_async_to_lds_b128((AS1 v4i_*)g, (AS3 v4i_*)l, 0, 0);
    };
#else
    const int r8 = tid >> 3, cg = tid & 7;
    uint2 ua[4], ub[2];
    auto ldgAB = [&](int k0) {
#pragma unroll
        for (int p = 0; p < 4; p++)
            ua[p] = *(const uint2*)(Ao + (size_t)(mb + p * 32 + r8) * E_DIM + k0 + cg * 4);
#pragma unroll
        for (int p = 0; p < 2; p++)
            ub[p] = *(const uint2*)(W + (size_t)(nb + p * 32 + r8) * E_DIM + k0 + cg * 4);
    };
    auto cstAB = [&](int buf) {
#pragma unroll
        for (int p = 0; p < 4; p++)
            *(uint2*)&a_lds[buf][(p * 32 + r8) * LDH + cg * 4] = ua[p];
#pragma unroll
        for (int p = 0; p < 2; p++)
            *(uint2*)&b_lds[buf][(p * 32 + r8) * LDH + cg * 4] = ub[p];
    };
#endif

#if HAVE_ASYNC_LDS
    stageA(0, 0); stageB(0, 0);
#else
    ldgAB(0); cstAB(0);
#endif

    for (int kt = 0; kt < 16; kt++) {
        const int cur = kt & 1;
        const bool hn = kt < 15;
#if HAVE_ASYNC_LDS
        wait_async0();
        __syncthreads();
        if (hn) { stageA(cur ^ 1, (kt + 1) * 32); stageB(cur ^ 1, (kt + 1) * 32); }
#else
        if (hn) ldgAB((kt + 1) * 32);
        __syncthreads();
#endif

        Frag af[2], bf_[2];
#pragma unroll
        for (int a = 0; a < 2; a++)
#pragma unroll
            for (int i = 0; i < 8; i++)
                af[a].u[i] = *(const u32*)&a_lds[cur][(wm + a * 16 + ln) * LDH + akmap(i, half)];
#pragma unroll
        for (int b = 0; b < 2; b++)
#pragma unroll
            for (int i = 0; i < 8; i++)
                bf_[b].u[i] = *(const u32*)&b_lds[cur][(wn + b * 16 + ln) * LDH + half * 16 + 2 * i];
#pragma unroll
        for (int a = 0; a < 2; a++)
#pragma unroll
            for (int b = 0; b < 2; b++)
                acc[a][b] = wmma_bf16(af[a], bf_[b], acc[a][b]);

#if !HAVE_ASYNC_LDS
        if (hn) cstAB(cur ^ 1);
#endif
    }

#pragma unroll
    for (int b = 0; b < 2; b++) {
        int nn = nb + wn + b * 16 + ln;
        float bv = bias[nn];
#pragma unroll
        for (int a = 0; a < 2; a++) {
#pragma unroll
            for (int i = 0; i < 8; i++) {
                int m = mb + wm + a * 16 + half * 8 + i;
                out[(size_t)m * E_DIM + nn] = acc[a][b][i] + bv;
            }
        }
    }
}

// ---------------------------------------------------------------------------
extern "C" void kernel_launch(void* const* d_in, const int* in_sizes, int n_in,
                              void* d_out, int out_size, void* d_ws, size_t ws_size,
                              hipStream_t stream) {
    const float* query = (const float*)d_in[0];   // [T,B,E]
    const float* dist  = (const float*)d_in[1];   // [B,T,T,H]
    const float* w_in  = (const float*)d_in[2];   // [3E,E]
    const float* b_in  = (const float*)d_in[3];   // [3E]
    const float* w_out = (const float*)d_in[4];   // [E,E]
    const float* b_out = (const float*)d_in[5];   // [E]
    float* out = (float*)d_out;                   // [T,B,E]

    char* ws = (char*)d_ws;
    u16* q_ws  = (u16*)(ws);                       // [BH][T][HD] bf16, 4 MB
    u16* k_ws  = (u16*)(ws + ( 4u << 20));         // [BH][T][HD] bf16, 4 MB
    u16* vT_ws = (u16*)(ws + ( 8u << 20));         // [BH][HD][T] bf16, 4 MB
    u16* o_ws  = (u16*)(ws + (12u << 20));         // [M][E]      bf16, 4 MB
    u16* q_bf  = (u16*)(ws + (16u << 20));         // query bf16, 4 MB
    u16* wi_bf = (u16*)(ws + (20u << 20));         // W_in  bf16, 1.5 MB
    u16* wo_bf = (u16*)(ws + (20u << 20) + 0x180000u);  // W_out bf16, 0.5 MB

    // pre-convert GEMM operands to bf16 (tiny vs. the 128MB dist stream)
    cvt_bf16_kernel<<<(M_DIM * E_DIM / 4) / 256, 256, 0, stream>>>(query, q_bf,
                                                                   M_DIM * E_DIM / 4);
    cvt_bf16_kernel<<<(3 * E_DIM * E_DIM / 4) / 256, 256, 0, stream>>>(w_in, wi_bf,
                                                                       3 * E_DIM * E_DIM / 4);
    cvt_bf16_kernel<<<(E_DIM * E_DIM / 4) / 256, 256, 0, stream>>>(w_out, wo_bf,
                                                                   E_DIM * E_DIM / 4);

    qkv_proj_kernel<<<dim3(M_DIM / 128, (3 * E_DIM) / 64), 256, 0, stream>>>(
        q_bf, wi_bf, b_in, q_ws, k_ws, vT_ws);
    attn_kernel<<<dim3(T_DIM / 128, BH_DIM), 256, 0, stream>>>(
        q_ws, k_ws, vT_ws, dist, o_ws);
    out_proj_kernel<<<dim3(M_DIM / 128, E_DIM / 64), 256, 0, stream>>>(
        o_ws, wo_bf, b_out, out);
}
